// CGAMSGNet_56624848831043
// MI455X (gfx1250) — compile-verified, matrix operand order checked by reading, not run on hardware
//
#include <hip/hip_runtime.h>
#include <hip/hip_bf16.h>
#include <math.h>

typedef _Float16 v16h __attribute__((ext_vector_type(16)));
typedef float    v8f  __attribute__((ext_vector_type(8)));

// ---------------------------------------------------------------------------
// Kernel 1: pairwise-distance top-128 per point (xyz channels only).
// ---------------------------------------------------------------------------
__global__ __launch_bounds__(256)
void cga_topk_kernel(const float* __restrict__ x, int* __restrict__ idx, int N)
{
    const int p   = blockIdx.x;       // b*N + n
    const int b   = p / N;
    const int n   = p % N;
    const int tid = threadIdx.x;

    __shared__ float dist[4096];
    __shared__ float rv[256];
    __shared__ int   ri[256];

    const float* xb = x + (size_t)b * 10 * N;
    const float cx = xb[0 * N + n], cy = xb[1 * N + n], cz = xb[2 * N + n];
    const float cc = cx * cx + cy * cy + cz * cz;

    for (int m = tid; m < N; m += 256) {
        __builtin_prefetch(&xb[m + 256], 0, 0);
        float mx = xb[m], my = xb[N + m], mz = xb[2 * N + m];
        dist[m] = 2.f * (cx * mx + cy * my + cz * mz) - cc - (mx * mx + my * my + mz * mz);
    }
    __syncthreads();

    for (int iter = 0; iter < 128; ++iter) {
        float bv = -3.0e38f; int bi = N;
        for (int m = tid; m < N; m += 256) {
            float v = dist[m];
            if (v > bv || (v == bv && m < bi)) { bv = v; bi = m; }
        }
        rv[tid] = bv; ri[tid] = bi;
        __syncthreads();
        for (int s = 128; s > 0; s >>= 1) {
            if (tid < s) {
                if (rv[tid + s] > rv[tid] ||
                    (rv[tid + s] == rv[tid] && ri[tid + s] < ri[tid])) {
                    rv[tid] = rv[tid + s]; ri[tid] = ri[tid + s];
                }
            }
            __syncthreads();
        }
        if (tid == 0) { idx[(size_t)p * 128 + iter] = ri[0]; dist[ri[0]] = -3.0e38f; }
        __syncthreads();
    }
}

// ---------------------------------------------------------------------------
// Kernel 2: edge conv via WMMA. Weight matrix stored N-major (wmatT[o*32+c])
// so both A and B fragments load as two contiguous 16B chunks (ds_load_b128).
// ---------------------------------------------------------------------------
__global__ __launch_bounds__(128)
void cga_edge_kernel(const float* __restrict__ x, const int* __restrict__ idx,
                     const float* __restrict__ w, int kk,
                     float* __restrict__ hmax,
                     float* __restrict__ M1, float* __restrict__ M2, int N)
{
    const int p    = blockIdx.x;      // b*N + n
    const int b    = p / N;
    const int n    = p % N;
    const int tid  = threadIdx.x;
    const int lane = tid & 31;
    const int wave = tid >> 5;

    __shared__ _Float16 fmat[128 * 32];   // k x Kpad feature rows (A)
    __shared__ _Float16 wmatT[64 * 32];   // o x Kpad weights     (B, N-major)
    __shared__ float    nbr[128 * 10];
    __shared__ float    hpart[8 * 64];
    __shared__ float    ctrS[10];
    __shared__ float    stdS[3];

    const float* xb = x + (size_t)b * 10 * N;
    if (tid < 10) ctrS[tid] = xb[tid * N + n];

    for (int i = tid; i < 64 * 32; i += 128) {
        int o = i >> 5, c = i & 31;
        wmatT[i] = (c < 10) ? (_Float16)w[o * 10 + c] : (_Float16)0.f;
    }
    if (tid < kk) {
        int ii = idx[(size_t)p * 128 + tid];
        for (int c = 0; c < 10; ++c) nbr[tid * 10 + c] = xb[c * N + ii];
    }
    __syncthreads();

    // std of neighbor rgb over k, ddof=1
    if (tid < 3) {
        float s = 0.f, s2 = 0.f;
        for (int j = 0; j < kk; ++j) { float v = nbr[j * 10 + 3 + tid]; s += v; s2 += v * v; }
        float var = (s2 - s * s / (float)kk) / (float)(kk - 1);
        stdS[tid] = sqrtf(fmaxf(var, 0.f));
    }
    __syncthreads();

    // feature rows: [delta_rgb(3) | std_rgb(3) | delta_xyz(3) | theta(1) | 0..]
    if (tid < kk) {
        const float* nb = &nbr[tid * 10];
        _Float16* row = &fmat[tid * 32];
        row[0] = (_Float16)(ctrS[3] - nb[3]);
        row[1] = (_Float16)(ctrS[4] - nb[4]);
        row[2] = (_Float16)(ctrS[5] - nb[5]);
        row[3] = (_Float16)stdS[0];
        row[4] = (_Float16)stdS[1];
        row[5] = (_Float16)stdS[2];
        row[6] = (_Float16)(ctrS[0] - nb[0]);
        row[7] = (_Float16)(ctrS[1] - nb[1]);
        row[8] = (_Float16)(ctrS[2] - nb[2]);
        row[9] = (_Float16)ctrS[6];
        for (int c = 10; c < 32; ++c) row[c] = (_Float16)0.f;
    }
    __syncthreads();

    // feature moments for BN stats: mean_h = W*M1, E[h^2] = diag(W*M2*W^T)
    if (tid < 10) {
        float s = 0.f;
        for (int j = 0; j < kk; ++j) s += (float)fmat[j * 32 + tid];
        atomicAdd(&M1[tid], s);
    } else if (tid >= 16 && tid < 116) {
        int u = tid - 16, c = u / 10, d = u % 10;
        float s = 0.f;
        for (int j = 0; j < kk; ++j)
            s += (float)fmat[j * 32 + c] * (float)fmat[j * 32 + d];
        atomicAdd(&M2[c * 10 + d], s);
    }

    // WMMA tiles: mtiles = k/16 along edges, 4 tiles along the 64 outputs
    const int mtiles = kk >> 4;
    const int tiles  = mtiles * 4;
    const int kb     = (lane >> 4) * 8;
    for (int t = wave; t < tiles; t += 4) {
        int mt = t >> 2, nt = t & 3;
        int mrow = mt * 16 + (lane & 15);
        int ncol = nt * 16 + (lane & 15);
        const _Float16* ap = &fmat [mrow * 32 + kb];
        const _Float16* bp = &wmatT[ncol * 32 + kb];
        v16h a, bm;
        for (int h = 0; h < 16; ++h) {
            int o = (h & 7) + ((h >= 8) ? 16 : 0);
            a[h]  = ap[o];
            bm[h] = bp[o];
        }
        v8f acc = {};
        acc = __builtin_amdgcn_wmma_f32_16x16x32_f16(
                  false, a, false, bm, (short)0, acc, false, false);
        float mx = acc[0];
        for (int r = 1; r < 8; ++r) mx = fmaxf(mx, acc[r]);
        mx = fmaxf(mx, __shfl_xor(mx, 16, 32));     // combine lane halves (wave32)
        if (lane < 16) hpart[mt * 64 + ncol] = mx;
    }
    __syncthreads();

    if (tid < 64) {
        float mx = hpart[tid];
        for (int mt = 1; mt < mtiles; ++mt) mx = fmaxf(mx, hpart[mt * 64 + tid]);
        hmax[(size_t)p * 64 + tid] = mx;
    }
}

// ---------------------------------------------------------------------------
// Edge-branch BN stat finalize: st = [mean(64) | g*rsqrt(var+eps)(64) | beta(64)]
// ---------------------------------------------------------------------------
__global__ void cga_edgefin_kernel(const float* __restrict__ M1, const float* __restrict__ M2,
                                   const float* __restrict__ w, const float* __restrict__ g,
                                   const float* __restrict__ bta, float cnt,
                                   float* __restrict__ st)
{
    int o = threadIdx.x;
    if (o < 64) {
        float mean = 0.f;
        for (int c = 0; c < 10; ++c) mean += w[o * 10 + c] * M1[c];
        mean /= cnt;
        float ex2 = 0.f;
        for (int c = 0; c < 10; ++c)
            for (int d = 0; d < 10; ++d)
                ex2 += w[o * 10 + c] * w[o * 10 + d] * M2[c * 10 + d];
        ex2 /= cnt;
        float var = ex2 - mean * mean;
        st[o]       = mean;
        st[64 + o]  = g[o] * rsqrtf(var + 1e-5f);
        st[128 + o] = bta[o];
    }
}

// Generic per-channel BN finalize from sum/sumsq.
__global__ void cga_bnfin_kernel(const float* __restrict__ sum, const float* __restrict__ sumsq,
                                 const float* __restrict__ g, const float* __restrict__ bta,
                                 float cnt, float* __restrict__ st, int C)
{
    int t = blockIdx.x * blockDim.x + threadIdx.x;
    if (t < C) {
        float m   = sum[t] / cnt;
        float var = sumsq[t] / cnt - m * m;
        st[t]         = m;
        st[C + t]     = g[t] * rsqrtf(var + 1e-5f);
        st[2 * C + t] = bta[t];
    }
}

// ---------------------------------------------------------------------------
// Kernel 3: BN+lrelu both branches, attention MLP + softmax, fuse.
// ---------------------------------------------------------------------------
__global__ __launch_bounds__(128)
void cga_fuse_kernel(const float* __restrict__ hm, const float* __restrict__ hM,
                     const float* __restrict__ stm, const float* __restrict__ stM,
                     const float* __restrict__ aw1, const float* __restrict__ ab1,
                     const float* __restrict__ aw2, const float* __restrict__ ab2,
                     float* __restrict__ fused)
{
    const int p = blockIdx.x;
    const int t = threadIdx.x;

    __shared__ float xm[64], xM[64], comb[128], att[64], sm[2];

    if (t < 64) {
        float v = (hm[(size_t)p * 64 + t] - stm[t]) * stm[64 + t] + stm[128 + t];
        v = (v >= 0.f) ? v : 0.2f * v;
        xm[t] = v; comb[t] = v;
    } else {
        int o = t - 64;
        float v = (hM[(size_t)p * 64 + o] - stM[o]) * stM[64 + o] + stM[128 + o];
        v = (v >= 0.f) ? v : 0.2f * v;
        xM[o] = v; comb[64 + o] = v;
    }
    __syncthreads();

    if (t < 64) {
        float a = ab1[t];
        for (int c = 0; c < 128; ++c) a += aw1[t * 128 + c] * comb[c];
        att[t] = fmaxf(a, 0.f);
    }
    __syncthreads();

    if (t < 2) {
        float a = ab2[t];
        for (int c = 0; c < 64; ++c) a += aw2[t * 64 + c] * att[c];
        sm[t] = a;
    }
    __syncthreads();
    if (t == 0) {
        float m = fmaxf(sm[0], sm[1]);
        float e0 = expf(sm[0] - m), e1 = expf(sm[1] - m);
        float s = e0 + e1;
        sm[0] = e0 / s; sm[1] = e1 / s;
    }
    __syncthreads();

    if (t < 64)
        fused[(size_t)p * 64 + t] = sm[0] * xm[t] + sm[1] * xM[t];
}

// ---------------------------------------------------------------------------
// WMMA linear layer over 16-point tiles:
//   hin = transform(yin)  (mode 0: identity, 1: bn+lrelu, 2: bn+relu)
//   yout = hin @ W^T (+bias), optional per-channel sum/sumsq accumulation.
// in_dim/out_dim must be multiples of 32/16 (64 or 128 here).
// ---------------------------------------------------------------------------
__global__ __launch_bounds__(128)
void cga_wmma_linear_kernel(const float* __restrict__ yin, const float* __restrict__ st,
                            const float* __restrict__ W, const float* __restrict__ bias,
                            float* __restrict__ yout, float* __restrict__ sum,
                            float* __restrict__ sq,
                            int in_dim, int out_dim, int mode)
{
    const int p0   = blockIdx.x * 16;
    const int tid  = threadIdx.x;
    const int lane = tid & 31;
    const int wave = tid >> 5;

    __shared__ _Float16 amat[16 * 128];     // 16 points x in_dim (A)
    __shared__ _Float16 wt[128 * 128];      // out_dim x in_dim   (B, N-major)

    for (int i = tid; i < 16 * in_dim; i += 128) {
        int m = i / in_dim, c = i - m * in_dim;
        float v = yin[(size_t)(p0 + m) * in_dim + c];
        if (mode) {
            v = (v - st[c]) * st[in_dim + c] + st[2 * in_dim + c];
            v = (mode == 2) ? fmaxf(v, 0.f) : ((v >= 0.f) ? v : 0.2f * v);
        }
        amat[i] = (_Float16)v;
    }
    for (int i = tid; i < out_dim * in_dim; i += 128)
        wt[i] = (_Float16)W[i];
    __syncthreads();

    const int ktiles = in_dim >> 5;
    const int ntiles = out_dim >> 4;
    const int kb     = (lane >> 4) * 8;
    const int mrow   = lane & 15;
    const int mbase  = (lane >> 4) * 8;

    for (int nt = wave; nt < ntiles; nt += 4) {
        const int ncol = nt * 16 + (lane & 15);
        v8f acc = {};
        for (int kt = 0; kt < ktiles; ++kt) {
            const _Float16* ap = &amat[mrow * in_dim + kt * 32 + kb];
            const _Float16* bp = &wt  [ncol * in_dim + kt * 32 + kb];
            v16h a, bm;
            for (int h = 0; h < 16; ++h) {
                int o = (h & 7) + ((h >= 8) ? 16 : 0);
                a[h]  = ap[o];
                bm[h] = bp[o];
            }
            acc = __builtin_amdgcn_wmma_f32_16x16x32_f16(
                      false, a, false, bm, (short)0, acc, false, false);
        }
        const float bv = bias ? bias[ncol] : 0.f;
        float s = 0.f, s2 = 0.f;
        for (int r = 0; r < 8; ++r) {
            float v = acc[r] + bv;
            yout[(size_t)(p0 + mbase + r) * out_dim + ncol] = v;
            s += v; s2 += v * v;
        }
        if (sum) {
            s  += __shfl_xor(s,  16, 32);
            s2 += __shfl_xor(s2, 16, 32);
            if (lane < 16) { atomicAdd(&sum[ncol], s); atomicAdd(&sq[ncol], s2); }
        }
    }
}

// ---------------------------------------------------------------------------
// Tiny final layer: h = relu(bn(yin)); yout = W@h + bias  (out_dim = 2).
// ---------------------------------------------------------------------------
__global__ __launch_bounds__(128)
void cga_layer_kernel(const float* __restrict__ yin, const float* __restrict__ st,
                      const float* __restrict__ W, const float* __restrict__ bias,
                      float* __restrict__ yout, int in_dim, int out_dim)
{
    const int p = blockIdx.x;
    const int t = threadIdx.x;
    __shared__ float hin[128];

    if (t < in_dim) {
        float v = (yin[(size_t)p * in_dim + t] - st[t]) * st[in_dim + t] + st[2 * in_dim + t];
        hin[t] = fmaxf(v, 0.f);
    }
    __syncthreads();

    if (t < out_dim) {
        float a = bias ? bias[t] : 0.f;
        for (int c = 0; c < in_dim; ++c) a += W[t * in_dim + c] * hin[c];
        yout[(size_t)p * out_dim + t] = a;
    }
}

// (B,N,2) -> (B,2,N)
__global__ void cga_transpose_kernel(const float* __restrict__ y4, float* __restrict__ out,
                                     int B, int N)
{
    int i = blockIdx.x * blockDim.x + threadIdx.x;
    int total = B * 2 * N;
    if (i < total) {
        int n = i % N;
        int c = (i / N) % 2;
        int b = i / (2 * N);
        out[i] = y4[((size_t)b * N + n) * 2 + c];
    }
}

// ---------------------------------------------------------------------------
extern "C" void kernel_launch(void* const* d_in, const int* in_sizes, int n_in,
                              void* d_out, int out_size, void* d_ws, size_t ws_size,
                              hipStream_t stream)
{
    (void)n_in; (void)out_size; (void)ws_size;

    const float* x       = (const float*)d_in[0];
    const float* w_micro = (const float*)d_in[1];
    const float* g_micro = (const float*)d_in[2];
    const float* b_micro = (const float*)d_in[3];
    const float* w_macro = (const float*)d_in[4];
    const float* g_macro = (const float*)d_in[5];
    const float* b_macro = (const float*)d_in[6];
    const float* aw1     = (const float*)d_in[7];
    const float* ab1     = (const float*)d_in[8];
    const float* aw2     = (const float*)d_in[9];
    const float* ab2     = (const float*)d_in[10];
    const float* tw1     = (const float*)d_in[11];
    const float* tg1     = (const float*)d_in[12];
    const float* tb1     = (const float*)d_in[13];
    const float* tw2     = (const float*)d_in[14];
    const float* tg2     = (const float*)d_in[15];
    const float* tb2     = (const float*)d_in[16];
    const float* cw1     = (const float*)d_in[17];
    const float* cb1     = (const float*)d_in[18];
    const float* cg      = (const float*)d_in[19];
    const float* cbeta   = (const float*)d_in[20];
    const float* cw2     = (const float*)d_in[21];
    const float* cb2     = (const float*)d_in[22];

    const int B = 2, C = 10;
    const int N = in_sizes[0] / (B * C);
    const int P = B * N;

    // ---- workspace carve-up (256B aligned) ----
    char*  base = (char*)d_ws;
    size_t off  = 0;
    auto alloc = [&](size_t bytes) -> char* {
        char* r = base + off;
        off = (off + bytes + 255) & ~(size_t)255;
        return r;
    };

    // accumulators, contiguous for a single memset: 768 floats
    float* acc  = (float*)alloc(768 * sizeof(float));
    float* M1m  = acc;          float* M2m = acc + 16;    // 16..115
    float* M1M  = acc + 128;    float* M2M = acc + 144;   // 144..243
    float* sum1 = acc + 256;    float* sq1 = acc + 384;   // 128 each
    float* sum2 = acc + 512;    float* sq2 = acc + 576;   // 64 each
    float* sum3 = acc + 640;    float* sq3 = acc + 704;   // 64 each

    float* stm  = (float*)alloc(192 * sizeof(float));
    float* stM  = (float*)alloc(192 * sizeof(float));
    float* st1  = (float*)alloc(384 * sizeof(float));
    float* st2  = (float*)alloc(192 * sizeof(float));
    float* st3  = (float*)alloc(192 * sizeof(float));

    int*   idx   = (int*)  alloc((size_t)P * 128 * sizeof(int));
    float* hm    = (float*)alloc((size_t)P * 64  * sizeof(float));
    float* hM    = (float*)alloc((size_t)P * 64  * sizeof(float));
    float* fused = (float*)alloc((size_t)P * 64  * sizeof(float));
    float* y1    = (float*)alloc((size_t)P * 128 * sizeof(float));
    float* y2    = (float*)alloc((size_t)P * 64  * sizeof(float));
    float* y3    = (float*)alloc((size_t)P * 64  * sizeof(float));
    float* y4    = (float*)alloc((size_t)P * 2   * sizeof(float));

    hipMemsetAsync(acc, 0, 768 * sizeof(float), stream);

    // 1) knn top-128 (micro branch uses the first 32 = top-32 set)
    cga_topk_kernel<<<P, 256, 0, stream>>>(x, idx, N);

    // 2) edge convs (WMMA) + feature moments
    cga_edge_kernel<<<P, 128, 0, stream>>>(x, idx, w_micro, 32,  hm, M1m, M2m, N);
    cga_edge_kernel<<<P, 128, 0, stream>>>(x, idx, w_macro, 128, hM, M1M, M2M, N);

    // 3) edge BN stats from feature moments
    cga_edgefin_kernel<<<1, 64, 0, stream>>>(M1m, M2m, w_micro, g_micro, b_micro,
                                             (float)((double)P * 32.0),  stm);
    cga_edgefin_kernel<<<1, 64, 0, stream>>>(M1M, M2M, w_macro, g_macro, b_macro,
                                             (float)((double)P * 128.0), stM);

    // 4) attention fuse
    cga_fuse_kernel<<<P, 128, 0, stream>>>(hm, hM, stm, stM, aw1, ab1, aw2, ab2, fused);

    // 5) trunk via WMMA linear layers (16-point tiles)
    const int PT = P / 16;
    cga_wmma_linear_kernel<<<PT, 128, 0, stream>>>(fused, nullptr, tw1, nullptr,
                                                   y1, sum1, sq1, 64, 128, 0);
    cga_bnfin_kernel<<<1, 128, 0, stream>>>(sum1, sq1, tg1, tb1, (float)P, st1, 128);

    cga_wmma_linear_kernel<<<PT, 128, 0, stream>>>(y1, st1, tw2, nullptr,
                                                   y2, sum2, sq2, 128, 64, 1);
    cga_bnfin_kernel<<<1, 64, 0, stream>>>(sum2, sq2, tg2, tb2, (float)P, st2, 64);

    cga_wmma_linear_kernel<<<PT, 128, 0, stream>>>(y2, st2, cw1, cb1,
                                                   y3, sum3, sq3, 64, 64, 1);
    cga_bnfin_kernel<<<1, 64, 0, stream>>>(sum3, sq3, cg, cbeta, (float)P, st3, 64);

    // 6) output layer: h3=relu(bn(y3)); y4 = h3 @ cw2^T + cb2
    cga_layer_kernel<<<P, 128, 0, stream>>>(y3, st3, cw2, cb2, y4, 64, 2);

    // 7) (B,N,2) -> (B,2,N)
    cga_transpose_kernel<<<(B * 2 * N + 255) / 256, 256, 0, stream>>>(y4, (float*)d_out, B, N);
}
